// SoftRoiPool_23691039604814
// MI455X (gfx1250) — compile-verified, matrix operand order checked by reading, not run on hardware
//
#include <hip/hip_runtime.h>

// ---------------------------------------------------------------------------
// SoftRoiPool for MI455X (gfx1250, wave32).
//
// Roofline: GEMM is 8.6 GFLOP over ~400 MB of mandatory traffic -> memory
// bound (~17 us @ 23.3 TB/s) when using V_WMMA_F32_16X16X32_F16 (f32
// accumulate). f32 WMMA (K=4) would be compute bound. map values are in
// [0,1], x ~ N(0,1): f16 inputs are fine. Async/TDM copies cannot convert
// f32->f16 nor produce the WMMA fragment swizzle, so data is staged through
// VGPRs into double-buffered LDS. x (256 MiB, read-once, > 192 MB L2) is
// loaded non-temporally so the 64 MiB roi_map stays L2-resident between
// kernel A (producer) and kernel B (consumer).
// ---------------------------------------------------------------------------

typedef _Float16 half_t;
typedef __attribute__((ext_vector_type(16))) _Float16 v16h;
typedef __attribute__((ext_vector_type(4)))  _Float16 v4h;
typedef __attribute__((ext_vector_type(8)))  float    v8f;
typedef __attribute__((ext_vector_type(4)))  float    v4f;

#define N_ 16
#define R_ 64
#define H_ 128
#define W_ 128
#define D_ 256
#define F_ (H_ * W_)          // 16384

#define KCHUNKS 16
#define KCHUNK  (F_ / KCHUNKS) // 1024
#define KSTEP   32
#define NITER   (KCHUNK / KSTEP) // 32

// ---------------------------------------------------------------------------
// Kernel A: roi_map (separable: max_{h,w} e^{ly+lx} = max_h e^ly * max_w e^lx),
// roi_params passthrough, and zero-init of roi_features for atomic accumulate.
// One block per (n,r); 128 threads (one per grid coordinate).
// ---------------------------------------------------------------------------
__global__ __launch_bounds__(128) void roi_map_kernel(
    const float* __restrict__ params,
    float* __restrict__ feat_out,     // [N,R,D]   zeroed here
    float* __restrict__ params_out,   // [N,R,4]   copy
    float* __restrict__ map_out)      // [N,R,H,W]
{
  __shared__ float spx[W_];
  __shared__ float spy[H_];
  __shared__ float red[128];

  const int t = threadIdx.x;
  const int b = blockIdx.x;                 // n*R + r
  const float* p = params + b * 4;
  const float mux = p[0], muy = p[1], sgx = p[2], sgy = p[3];

  const float gx = (t + 0.5f) / (float)W_;
  const float gy = (t + 0.5f) / (float)H_;
  const float dx = fabsf(gx - mux) / sgx;
  const float dy = fabsf(gy - muy) / sgy;
  const float px = __expf(-dx * dx);        // BETA = 2
  const float py = __expf(-dy * dy);

  // max over w of px
  red[t] = px; __syncthreads();
  for (int s = 64; s > 0; s >>= 1) {
    if (t < s) red[t] = fmaxf(red[t], red[t + s]);
    __syncthreads();
  }
  const float maxx = red[0];
  __syncthreads();
  // max over h of py
  red[t] = py; __syncthreads();
  for (int s = 64; s > 0; s >>= 1) {
    if (t < s) red[t] = fmaxf(red[t], red[t + s]);
    __syncthreads();
  }
  const float maxy = red[0];

  const float inv = 1.0f / (maxx * maxy + 1e-12f);
  spx[t] = px * inv;
  spy[t] = py;
  __syncthreads();

  // Write roi_map with float4 stores (regular temporal hint: kernel B
  // re-reads it and it fits in the 192 MB L2).
  float* mrow = map_out + (size_t)b * F_;
  const int c4 = (t & 31) * 4;
  const float x0 = spx[c4 + 0], x1 = spx[c4 + 1], x2 = spx[c4 + 2], x3 = spx[c4 + 3];
  for (int it = 0; it < 32; ++it) {
    const int h = it * 4 + (t >> 5);
    const float yv = spy[h];
    v4f o = { x0 * yv, x1 * yv, x2 * yv, x3 * yv };
    *(v4f*)(mrow + (size_t)h * W_ + c4) = o;
  }

  // roi_params passthrough
  if (t < 4) params_out[b * 4 + t] = p[t];

  // zero roi_features[b, :] for atomic accumulation in kernel B
  if (t < 64) {
    v4f z = { 0.f, 0.f, 0.f, 0.f };
    *(v4f*)(feat_out + (size_t)b * D_ + t * 4) = z;
  }
}

// ---------------------------------------------------------------------------
// Kernel B: batched GEMM  feat[n,r,d] += sum_f map[n,r,f] * x[n,f,d] / F
// Grid: 16 K-chunks x 16 batches (x/map each read exactly once globally).
// Block: 512 threads = 16 waves. Wave (m,g): M-tile m (16 rows of r),
// col group g (4 N-tiles of 16 cols).
//
// Software pipeline (one barrier per K-step, double-buffered LDS):
//   - issue non-temporal global float4 loads of x for step it+1 into
//     registers (they fly under the WMMAs)
//   - 4x v_wmma_f32_16x16x32_f16 from buf[it&1] (contiguous ds_load_b128
//     fragments; LDS holds data pre-swizzled into the WMMA per-lane layout:
//     lane<16: K {0..7,16..23}; lane>=16: K {8..15,24..31})
//   - convert f32->f16 (v_cvt_pk_f16_f32 pairs) and store into buf[(it+1)&1]
//     with WIDE LDS stores: each thread stages a 4k x 4col sub-block, and an
//     aligned run of 4 K values is contiguous in j within one lane's
//     fragment -> 4x ds_store_b64 per thread (vs 16x ds_store_b16)
//   - __syncthreads()
// Epilogue: scale by 1/F, relaxed agent-scope f32 atomic adds merge the
// 16 K-chunk partials (~16 MB of atomics, negligible vs 400 MB streams).
// ---------------------------------------------------------------------------
__global__ __launch_bounds__(512) void roi_gemm_kernel(
    const float* __restrict__ x,      // [N,F,D]
    const float* __restrict__ map,    // [N,R,F]
    float* __restrict__ feat)         // [N,R,D]
{
  __shared__ __align__(16) half_t Abuf[2][4 * 32 * 16];   // [buf][mtile][lane][j]  2x4 KB
  __shared__ __align__(16) half_t Bbuf[2][16 * 32 * 16];  // [buf][ntile][lane][j]  2x16 KB

  const int t    = threadIdx.x;
  const int bx   = blockIdx.x;
  const int kc   = bx & (KCHUNKS - 1);
  const int n    = bx >> 4;
  const int lane = t & 31;
  const int wv   = t >> 5;
  const int m    = wv & 3;   // M tile (rows m*16 .. m*16+15)
  const int g    = wv >> 2;  // column group (cols g*64 .. g*64+63)

  v8f acc[4] = {};

  // --- A loader indices: thread t loads map row (t>>3), 4 K at (t&7)*4 ---
  const int arow = t >> 3;              // 0..63
  const int akq  = (t & 7) * 4;         // 0,4,...,28
  const int ahi  = (akq >> 3) & 1;
  const int aj   = (akq & 7) + ((akq >> 4) & 1) * 8;   // contiguous run of 4
  const int aidx = (((arow >> 4) * 32) + (arow & 15) + ahi * 16) * 16 + aj;
  const float* aptr = map + (size_t)(n * R_ + arow) * F_ + kc * KCHUNK + akq;

  // --- B loader indices: thread t stages a 4k x 4col sub-block:
  //     k in [bkq, bkq+4), cols [cb, cb+4). Per-column, the 4 K values are
  //     contiguous j slots of one lane -> one ds_store_b64 per column. ---
  const int bkq = (t >> 6) * 4;         // 0,4,...,28
  const int cb  = (t & 63) * 4;         // 0..252
  const int bhi = (bkq >> 3) & 1;
  const int bj  = (bkq & 7) + ((bkq >> 4) & 1) * 8;    // contiguous run of 4
  int bidx[4];
#pragma unroll
  for (int i = 0; i < 4; ++i) {
    const int col = cb + i;
    bidx[i] = (col >> 4) * 512 + ((col & 15) + bhi * 16) * 16 + bj;
  }
  const float* bptr = x + ((size_t)n * F_ + (size_t)(kc * KCHUNK + bkq)) * D_ + cb;

  // ---- prologue: stage K-step 0 into buffer 0 ----
  v4f areg = *(const v4f*)(aptr);
  v4f breg[4];                          // breg[kk] = x[bkq+kk][cb..cb+3]
#pragma unroll
  for (int kk = 0; kk < 4; ++kk)
    breg[kk] = __builtin_nontemporal_load((const v4f*)(bptr + (size_t)kk * D_));
  {
    v4h ah = { (half_t)areg[0], (half_t)areg[1], (half_t)areg[2], (half_t)areg[3] };
    *(v4h*)&Abuf[0][aidx] = ah;
#pragma unroll
    for (int i = 0; i < 4; ++i) {
      v4h bh = { (half_t)breg[0][i], (half_t)breg[1][i],
                 (half_t)breg[2][i], (half_t)breg[3][i] };
      *(v4h*)&Bbuf[0][bidx[i]] = bh;
    }
  }
  __syncthreads();

  for (int it = 0; it < NITER; ++it) {
    const int cur = it & 1;
    const int nxt = cur ^ 1;

    // ---- register prefetch for step it+1 (overlaps with WMMA below) ----
    const bool more = (it + 1 < NITER);
    if (more) {
      const float* ap = aptr + (it + 1) * KSTEP;
      const float* bp = bptr + (size_t)(it + 1) * KSTEP * D_;
      areg = *(const v4f*)ap;
#pragma unroll
      for (int kk = 0; kk < 4; ++kk)
        breg[kk] = __builtin_nontemporal_load((const v4f*)(bp + (size_t)kk * D_));
      if (it + 2 < NITER)
        __builtin_prefetch(bp + (size_t)KSTEP * D_, 0, 0);  // global_prefetch_b8
    }

    // ---- WMMA on current buffer: contiguous 32-B per-lane fragments ----
    v16h a = *(const v16h*)&Abuf[cur][(m * 32 + lane) * 16];
#pragma unroll
    for (int s = 0; s < 4; ++s) {
      v16h b = *(const v16h*)&Bbuf[cur][((g * 4 + s) * 32 + lane) * 16];
      acc[s] = __builtin_amdgcn_wmma_f32_16x16x32_f16(
          /*neg_a=*/false, a, /*neg_b=*/false, b,
          /*c_mod=*/(short)0, acc[s], /*reuse_a=*/false, /*reuse_b=*/false);
    }

    // ---- convert + store prefetched registers into the other buffer ----
    if (more) {
      v4h ah = { (half_t)areg[0], (half_t)areg[1], (half_t)areg[2], (half_t)areg[3] };
      *(v4h*)&Abuf[nxt][aidx] = ah;
#pragma unroll
      for (int i = 0; i < 4; ++i) {
        v4h bh = { (half_t)breg[0][i], (half_t)breg[1][i],
                   (half_t)breg[2][i], (half_t)breg[3][i] };
        *(v4h*)&Bbuf[nxt][bidx[i]] = bh;
      }
    }

    // single barrier per K-step: publishes buf[nxt] for step it+1 and
    // separates this step's reads of buf[cur] from its overwrite at it+2
    __syncthreads();
  }

  // ---- epilogue: C/D layout: VGPR j -> M = j (+8 for lanes >= 16), N = lane&15
  const float scale = 1.0f / (float)F_;
  const int colb = g * 64 + (lane & 15);
  const int mb   = m * 16 + ((lane >> 4) << 3);
#pragma unroll
  for (int s = 0; s < 4; ++s) {
#pragma unroll
    for (int j = 0; j < 8; ++j) {
      float v = acc[s][j] * scale;
      float* addr = feat + (size_t)(n * R_ + mb + j) * D_ + colb + s * 16;
      __hip_atomic_fetch_add(addr, v, __ATOMIC_RELAXED, __HIP_MEMORY_SCOPE_AGENT);
    }
  }
}

// ---------------------------------------------------------------------------
extern "C" void kernel_launch(void* const* d_in, const int* in_sizes, int n_in,
                              void* d_out, int out_size, void* d_ws, size_t ws_size,
                              hipStream_t stream) {
  const float* x          = (const float*)d_in[0];  // (16,128,128,256) f32
  const float* roi_params = (const float*)d_in[1];  // (16,64,4) f32

  float* out        = (float*)d_out;
  float* feat       = out;                                   // 16*64*256
  float* params_out = out + (size_t)N_ * R_ * D_;            // +262144
  float* map_out    = params_out + (size_t)N_ * R_ * 4;      // +4096

  roi_map_kernel<<<N_ * R_, 128, 0, stream>>>(roi_params, feat, params_out, map_out);
  roi_gemm_kernel<<<N_ * KCHUNKS, 512, 0, stream>>>(x, map_out, feat);
}
